// ContinuousEventTransformer_74448963109498
// MI455X (gfx1250) — matrix operand order repaired and checked by reading, not compiled, not run on hardware
//
#include <hip/hip_runtime.h>
#include <math.h>

// ---------------- problem constants (from reference) ----------------
static const int BB   = 2;
static const int TT   = 2048;
static const int INN  = 32;
static const int DD   = 512;
static const int HH   = 8;
static const int LL   = 6;
static const int METAN= 8;
static const int ACTN = 3;
static const int FFN  = 4 * DD;      // 2048
static const int HD   = DD / HH;     // 64
static const int MM   = BB * TT;     // 4096 rows
static const int S3   = 3 * DD;      // 1536

// ---------------- types ----------------
typedef __attribute__((ext_vector_type(16))) __bf16 bf16x16;
typedef __attribute__((ext_vector_type(8)))  float  f32x8;

__device__ __forceinline__ unsigned short f2bf(float f) {
    union { float f; unsigned int u; } c; c.f = f;
    unsigned int u = c.u;
    unsigned int r = (u + 0x7FFFu + ((u >> 16) & 1u)) >> 16;   // RNE
    return (unsigned short)r;
}

// A-operand fragment (16x32 bf16, row-major source):
// lane<16 holds row M=lane, K = kb..kb+15; lane>=16: kb += 16.
__device__ __forceinline__ bf16x16 load_a_frag(const unsigned short* __restrict__ A,
                                               int ldk, int row, int kbase) {
    union { uint4 u[2]; bf16x16 v; } f;
    const uint4* p = reinterpret_cast<const uint4*>(A + (size_t)row * ldk + kbase);
    f.u[0] = p[0];      // K = kbase .. kbase+7
    f.u[1] = p[1];      // K = kbase+8 .. kbase+15
    return f.v;
}

// 16 contiguous bf16 (packed B fragments / K-row fragments / V^T fragments)
__device__ __forceinline__ bf16x16 load_contig16(const unsigned short* __restrict__ p) {
    union { uint4 u[2]; bf16x16 v; } f;
    const uint4* q = reinterpret_cast<const uint4*>(p);
    f.u[0] = q[0]; f.u[1] = q[1];
    return f.v;
}

// ---------------- weight pre-pack into WMMA B-operand layout ----------------
// B (32x16 bf16) packing used here: lanes 0-15 hold K=0..15 of column N=lane,
// lanes 16-31 hold K=16..31 of column N=lane-16 (contiguous k order, same as load_a_frag; not the ISA table order).
__global__ void pack_weight_bf16(const float* __restrict__ W,
                                 unsigned short* __restrict__ Bp,
                                 int Kdim, int Ndim) {
    long idx = (long)blockIdx.x * blockDim.x + threadIdx.x;
    long total = (long)Kdim * Ndim;
    if (idx >= total) return;
    int  i    = (int)(idx & 15);
    int  lane = (int)((idx >> 4) & 31);
    long tile = idx >> 9;
    int  nTiles = Ndim >> 4;
    int  nt = (int)(tile % nTiles);
    int  kt = (int)(tile / nTiles);
    int  n  = nt * 16 + (lane & 15);
    int  k  = kt * 32 + (lane >> 4) * 16 + i;
    Bp[idx] = f2bf(W[(size_t)k * Ndim + n]);
}

// ---------------- V transpose: qkv V-part -> vT[B,H,HD,T] (keys contiguous) ----
__global__ void __launch_bounds__(256)
transpose_v(const unsigned short* __restrict__ qkvb, unsigned short* __restrict__ vT) {
    long idx = (long)blockIdx.x * 256 + threadIdx.x;   // over B*D*T = B*H*HD*T
    if (idx >= (long)BB * DD * TT) return;
    int  t  = (int)(idx & (TT - 1));
    long r  = idx >> 11;                // TT = 2^11
    int  d  = (int)(r & (HD - 1));
    long r2 = r >> 6;                   // HD = 2^6
    int  h  = (int)(r2 & (HH - 1));
    int  b  = (int)(r2 >> 3);           // HH = 2^3
    vT[idx] = qkvb[((size_t)b * TT + t) * S3 + 2 * DD + h * HD + d];
}

// ---------------- WMMA GEMM: C = act(A@W + bias) (+resid) ----------------
// A: bf16 [M,K] row-major.  Bp: packed bf16 fragments.
// Each wave computes NT 16x16 tiles (16 x NT*16 strip), A-fragment reused NT times.
// Grid must satisfy: nTiles == gridDim.x * wavesPerBlock * NT  (no guards ->
// uniform control flow, EXEC all-ones for WMMA).
template <int NT>
__global__ void __launch_bounds__(128)
gemm_bf16_wmma(const unsigned short* __restrict__ A,
               const unsigned short* __restrict__ Bp,
               const float* __restrict__ bias,
               const float* __restrict__ resid,
               float* __restrict__ outF,
               unsigned short* __restrict__ outB,
               int Ndim, int Kdim, int act) {
    int lane = threadIdx.x & 31;
    int wave = threadIdx.x >> 5;
    int wavesPerBlock = blockDim.x >> 5;
    int nTiles = Ndim >> 4;
    int nt0 = (blockIdx.x * wavesPerBlock + wave) * NT;
    int mt = blockIdx.y;
    int lr = lane & 15;
    int hb = lane >> 4;

    f32x8 c[NT];
#pragma unroll
    for (int j = 0; j < NT; ++j) c[j] = (f32x8){0.f,0.f,0.f,0.f,0.f,0.f,0.f,0.f};

    int row = mt * 16 + lr;
    int ksteps = Kdim >> 5;
    for (int kt = 0; kt < ksteps; ++kt) {
        bf16x16 a = load_a_frag(A, Kdim, row, kt * 32 + hb * 16);
        const unsigned short* bp = Bp + ((long)(kt * nTiles + nt0) * 32 + lane) * 16;
#pragma unroll
        for (int j = 0; j < NT; ++j) {
            bf16x16 b = load_contig16(bp + (long)j * 512);   // next tile: 32 lanes * 16
            c[j] = __builtin_amdgcn_wmma_f32_16x16x32_bf16(false, a, false, b,
                                                           (short)0, c[j], false, false);
        }
    }
#pragma unroll
    for (int j = 0; j < NT; ++j) {
        int col = (nt0 + j) * 16 + lr;
        float bv = bias ? bias[col] : 0.f;
#pragma unroll
        for (int r = 0; r < 8; ++r) {
            int orow = mt * 16 + r + hb * 8;     // C layout: VGPR r -> M=r (+8 upper half)
            float v = c[j][r] + bv;
            if (act == 1) v = 0.5f * v * (1.0f + erff(v * 0.70710678118f)); // exact GELU
            if (resid) v += resid[(size_t)orow * Ndim + col];
            size_t o = (size_t)orow * Ndim + col;
            if (outF) outF[o] = v;
            if (outB) outB[o] = f2bf(v);
        }
    }
}

// ---------------- LayerNorm: bf16 (and optional f32) output ----------------
__global__ void __launch_bounds__(256)
ln_kernel(const float* __restrict__ x, const float* __restrict__ g,
          const float* __restrict__ b, unsigned short* __restrict__ outB,
          float* __restrict__ outF, int Dd) {
    __shared__ float red[256];
    int rowi = blockIdx.x;
    const float* xr = x + (size_t)rowi * Dd;
    float s = 0.f;
    for (int i = threadIdx.x; i < Dd; i += 256) s += xr[i];
    red[threadIdx.x] = s; __syncthreads();
    for (int off = 128; off > 0; off >>= 1) {
        if ((int)threadIdx.x < off) red[threadIdx.x] += red[threadIdx.x + off];
        __syncthreads();
    }
    float mean = red[0] / Dd;
    __syncthreads();
    float vs = 0.f;
    for (int i = threadIdx.x; i < Dd; i += 256) { float d = xr[i] - mean; vs += d * d; }
    red[threadIdx.x] = vs; __syncthreads();
    for (int off = 128; off > 0; off >>= 1) {
        if ((int)threadIdx.x < off) red[threadIdx.x] += red[threadIdx.x + off];
        __syncthreads();
    }
    float rstd = rsqrtf(red[0] / Dd + 1e-5f);
    for (int i = threadIdx.x; i < Dd; i += 256) {
        float v = (xr[i] - mean) * rstd * g[i] + b[i];
        size_t o = (size_t)rowi * Dd + i;
        if (outB) outB[o] = f2bf(v);
        if (outF) outF[o] = v;
    }
}

// ---------------- Embedding: h = LN(ev@W_in+b)*g+beta + pos + meta@W_meta+b ----------------
__global__ void __launch_bounds__(256)
embed_kernel(const float* __restrict__ events, const float* __restrict__ meta,
             const float* __restrict__ W_in, const float* __restrict__ b_in,
             const float* __restrict__ g_in, const float* __restrict__ beta_in,
             const float* __restrict__ pos_emb, const float* __restrict__ W_meta,
             const float* __restrict__ b_meta, float* __restrict__ h) {
    __shared__ float ev[INN];
    __shared__ float mm[METAN];
    __shared__ float y[DD];
    __shared__ float red[256];
    int row = blockIdx.x;                 // b*T + t
    int b = row / TT, t = row % TT;
    if ((int)threadIdx.x < INN)   ev[threadIdx.x] = events[(size_t)row * INN + threadIdx.x];
    if ((int)threadIdx.x < METAN) mm[threadIdx.x] = meta[b * METAN + threadIdx.x];
    __syncthreads();
    for (int d = threadIdx.x; d < DD; d += 256) {
        float acc = b_in[d];
#pragma unroll
        for (int i = 0; i < INN; ++i) acc += ev[i] * W_in[i * DD + d];
        y[d] = acc;
    }
    __syncthreads();
    float s = 0.f;
    for (int d = threadIdx.x; d < DD; d += 256) s += y[d];
    red[threadIdx.x] = s; __syncthreads();
    for (int off = 128; off > 0; off >>= 1) {
        if ((int)threadIdx.x < off) red[threadIdx.x] += red[threadIdx.x + off];
        __syncthreads();
    }
    float mean = red[0] / DD;
    __syncthreads();
    float vs = 0.f;
    for (int d = threadIdx.x; d < DD; d += 256) { float dv = y[d] - mean; vs += dv * dv; }
    red[threadIdx.x] = vs; __syncthreads();
    for (int off = 128; off > 0; off >>= 1) {
        if ((int)threadIdx.x < off) red[threadIdx.x] += red[threadIdx.x + off];
        __syncthreads();
    }
    float rstd = rsqrtf(red[0] / DD + 1e-5f);
    for (int d = threadIdx.x; d < DD; d += 256) {
        float mp = b_meta[d];
#pragma unroll
        for (int j = 0; j < METAN; ++j) mp += mm[j] * W_meta[j * DD + d];
        float v = (y[d] - mean) * rstd * g_in[d] + beta_in[d] + pos_emb[(size_t)t * DD + d] + mp;
        h[(size_t)row * DD + d] = v;
    }
}

// ---------------- Flash attention: 1 wave / 16-query tile / (b,h) ----------------
// vT: [B,H,HD,T] bf16 (keys contiguous) -> V fragments are 2x b128 loads.
__global__ void __launch_bounds__(32)
attn_kernel(const unsigned short* __restrict__ qkv,
            const unsigned short* __restrict__ vT,
            unsigned short* __restrict__ attb) {
    __shared__ unsigned short pls[16][34];
    const float SCALE = 0.125f;               // 1/sqrt(64)
    int lane = threadIdx.x;
    int lr = lane & 15;
    int hb = lane >> 4;
    int qt = blockIdx.x;
    int bh = blockIdx.y;
    int b = bh / HH, h = bh % HH;
    const size_t rowBase = (size_t)b * TT;
    const int kcol = DD + h * HD;
    const unsigned short* vTh = vT + ((size_t)(b * HH + h) * HD) * TT;

    // Q fragments (A-operand, 16x64 split as two 16x32)
    const unsigned short* Aq = qkv + rowBase * S3 + h * HD;
    bf16x16 aq0 = load_a_frag(Aq, S3, qt * 16 + lr, hb * 16);
    bf16x16 aq1 = load_a_frag(Aq, S3, qt * 16 + lr, 32 + hb * 16);

    float mrow[8], lrow[8];
    f32x8 o[4];
#pragma unroll
    for (int r = 0; r < 8; ++r) { mrow[r] = -__builtin_inff(); lrow[r] = 0.f; }
#pragma unroll
    for (int d = 0; d < 4; ++d) o[d] = (f32x8){0.f,0.f,0.f,0.f,0.f,0.f,0.f,0.f};

    int nch = (qt * 16 + 16 + 31) >> 5;       // causal: keys < (qt+1)*16
    for (int ch = 0; ch < nch; ++ch) {
        int kb0 = ch * 32;
        f32x8 st[2];
#pragma unroll
        for (int kt2 = 0; kt2 < 2; ++kt2) {
            int kb = kb0 + kt2 * 16;
            const unsigned short* kp = qkv + (rowBase + kb + lr) * S3 + kcol;
            bf16x16 bk0 = load_contig16(kp + hb * 16);        // d = 0..31 chunk
            bf16x16 bk1 = load_contig16(kp + 32 + hb * 16);   // d = 32..63 chunk
            f32x8 s = {0.f,0.f,0.f,0.f,0.f,0.f,0.f,0.f};
            s = __builtin_amdgcn_wmma_f32_16x16x32_bf16(false, aq0, false, bk0, (short)0, s, false, false);
            s = __builtin_amdgcn_wmma_f32_16x16x32_bf16(false, aq1, false, bk1, (short)0, s, false, false);
#pragma unroll
            for (int r = 0; r < 8; ++r) {
                int qrow = qt * 16 + r + hb * 8;
                int krow = kb + lr;
                s[r] = (krow <= qrow) ? s[r] * SCALE : -__builtin_inff();
            }
            st[kt2] = s;
        }
        // streaming softmax (rows split across lane halves; reduce within 16-lane group)
#pragma unroll
        for (int r = 0; r < 8; ++r) {
            float mx = fmaxf(st[0][r], st[1][r]);
            for (int off = 1; off < 16; off <<= 1) mx = fmaxf(mx, __shfl_xor(mx, off));
            float mn = fmaxf(mrow[r], mx);
            float al = __expf(mrow[r] - mn);                  // 0 when mrow=-inf
            float p0 = __expf(st[0][r] - mn);
            float p1 = __expf(st[1][r] - mn);
            st[0][r] = p0; st[1][r] = p1;
            float rs = p0 + p1;
            for (int off = 1; off < 16; off <<= 1) rs += __shfl_xor(rs, off);
            lrow[r] = lrow[r] * al + rs;
            mrow[r] = mn;
#pragma unroll
            for (int d = 0; d < 4; ++d) o[d][r] *= al;
        }
        // P: C-layout -> A-layout via LDS
        __syncthreads();
#pragma unroll
        for (int r = 0; r < 8; ++r) {
            pls[r + hb * 8][lr]      = f2bf(st[0][r]);
            pls[r + hb * 8][16 + lr] = f2bf(st[1][r]);
        }
        __syncthreads();
        bf16x16 ap;
        {
            union { unsigned short s[16]; bf16x16 v; } f;
            int base = hb * 16;
#pragma unroll
            for (int i = 0; i < 16; ++i) {
                int k = base + i;
                f.s[i] = pls[lr][k];
            }
            ap = f.v;
        }
        // O += P @ V : V^T fragments are contiguous (keys-major) -> 2x b128 each
#pragma unroll
        for (int dt = 0; dt < 4; ++dt) {
            const unsigned short* vp = vTh + ((size_t)(dt * 16 + lr)) * TT + kb0 + hb * 16;
            bf16x16 bv = load_contig16(vp);
            o[dt] = __builtin_amdgcn_wmma_f32_16x16x32_bf16(false, ap, false, bv,
                                                            (short)0, o[dt], false, false);
        }
    }
    // epilogue: O /= l, store bf16
#pragma unroll
    for (int r = 0; r < 8; ++r) {
        int orow = qt * 16 + r + hb * 8;
        float inv = 1.0f / lrow[r];
        size_t base = (rowBase + orow) * (size_t)DD + h * HD;
#pragma unroll
        for (int dt = 0; dt < 4; ++dt)
            attb[base + dt * 16 + lr] = f2bf(o[dt][r] * inv);
    }
}

// ---------------- tiny action head ----------------
__global__ void action_kernel(const float* __restrict__ hf, const float* __restrict__ W_act,
                              const float* __restrict__ b_act, float* __restrict__ out) {
    int idx = threadIdx.x;
    if (idx >= BB * ACTN) return;
    int b = idx / ACTN, a = idx % ACTN;
    const float* hr = hf + ((size_t)b * TT + (TT - 1)) * DD;
    float s = b_act[a];
    for (int d = 0; d < DD; ++d) s += hr[d] * W_act[d * ACTN + a];
    out[idx] = s;
}

// ---------------- orchestration ----------------
extern "C" void kernel_launch(void* const* d_in, const int* in_sizes, int n_in,
                              void* d_out, int out_size, void* d_ws, size_t ws_size,
                              hipStream_t stream) {
    (void)in_sizes; (void)n_in; (void)out_size; (void)ws_size;
    const float* events  = (const float*)d_in[0];
    const float* meta    = (const float*)d_in[1];
    const float* W_in    = (const float*)d_in[2];
    const float* b_in    = (const float*)d_in[3];
    const float* g_in    = (const float*)d_in[4];
    const float* beta_in = (const float*)d_in[5];
    const float* pos_emb = (const float*)d_in[6];
    const float* W_meta  = (const float*)d_in[7];
    const float* b_meta  = (const float*)d_in[8];
    const float* ln1_g   = (const float*)d_in[9];
    const float* ln1_b   = (const float*)d_in[10];
    const float* Wqkv    = (const float*)d_in[11];
    const float* bqkv    = (const float*)d_in[12];
    const float* Wproj   = (const float*)d_in[13];
    const float* bproj   = (const float*)d_in[14];
    const float* ln2_g   = (const float*)d_in[15];
    const float* ln2_b   = (const float*)d_in[16];
    const float* W1      = (const float*)d_in[17];
    const float* b1      = (const float*)d_in[18];
    const float* W2      = (const float*)d_in[19];
    const float* b2      = (const float*)d_in[20];
    const float* lnf_g   = (const float*)d_in[21];
    const float* lnf_b   = (const float*)d_in[22];
    const float* W_ne    = (const float*)d_in[23];
    const float* b_ne    = (const float*)d_in[24];
    const float* W_act   = (const float*)d_in[25];
    const float* b_act   = (const float*)d_in[26];

    // workspace layout
    char* w = (char*)d_ws;
    float* h            = (float*)w;          w += (size_t)MM * DD * 4;
    unsigned short* xb  = (unsigned short*)w; w += (size_t)MM * DD * 2;
    unsigned short* qkvb= (unsigned short*)w; w += (size_t)MM * S3 * 2;
    unsigned short* attb= (unsigned short*)w; w += (size_t)MM * DD * 2;
    unsigned short* ffb = (unsigned short*)w; w += (size_t)MM * FFN * 2;
    unsigned short* vT  = (unsigned short*)w; w += (size_t)BB * DD * TT * 2;
    unsigned short* WqkvP = (unsigned short*)w; w += (size_t)LL * DD * S3 * 2;
    unsigned short* WprojP= (unsigned short*)w; w += (size_t)LL * DD * DD * 2;
    unsigned short* W1P   = (unsigned short*)w; w += (size_t)LL * DD * FFN * 2;
    unsigned short* W2P   = (unsigned short*)w; w += (size_t)LL * FFN * DD * 2;
    unsigned short* WneP  = (unsigned short*)w; w += (size_t)DD * INN * 2;

    // pack all weights to WMMA B-fragment layout (bf16)
    for (int l = 0; l < LL; ++l) {
        long n;
        n = (long)DD * S3;
        pack_weight_bf16<<<dim3((n + 255) / 256), 256, 0, stream>>>(Wqkv + (size_t)l * n, WqkvP + (size_t)l * n, DD, S3);
        n = (long)DD * DD;
        pack_weight_bf16<<<dim3((n + 255) / 256), 256, 0, stream>>>(Wproj + (size_t)l * n, WprojP + (size_t)l * n, DD, DD);
        n = (long)DD * FFN;
        pack_weight_bf16<<<dim3((n + 255) / 256), 256, 0, stream>>>(W1 + (size_t)l * n, W1P + (size_t)l * n, DD, FFN);
        n = (long)FFN * DD;
        pack_weight_bf16<<<dim3((n + 255) / 256), 256, 0, stream>>>(W2 + (size_t)l * n, W2P + (size_t)l * n, FFN, DD);
    }
    {
        long n = (long)DD * INN;
        pack_weight_bf16<<<dim3((n + 255) / 256), 256, 0, stream>>>(W_ne, WneP, DD, INN);
    }

    // embedding
    embed_kernel<<<dim3(MM), 256, 0, stream>>>(events, meta, W_in, b_in, g_in, beta_in,
                                               pos_emb, W_meta, b_meta, h);

    // GEMM grids: nTiles must equal gridDim.x * wavesPerBlock * NT exactly.
    // NT=4, 4 waves/block -> 256 columns per block:
    //   QKV  N=1536 -> grid.x=6 ; proj/W2 N=512 -> grid.x=2 ; W1 N=2048 -> grid.x=8
    // Head N=32 -> NT=2, 1 wave/block, grid.x=1.
    dim3 gemmBlk(128);
    long nvT = (long)BB * DD * TT;
    for (int l = 0; l < LL; ++l) {
        // x = LN1(h) -> bf16
        ln_kernel<<<dim3(MM), 256, 0, stream>>>(h, ln1_g + l * DD, ln1_b + l * DD, xb, nullptr, DD);
        // qkv = x @ Wqkv + bqkv  (bf16 out for attention)
        gemm_bf16_wmma<4><<<dim3(S3 / 256, MM / 16), gemmBlk, 0, stream>>>(
            xb, WqkvP + (size_t)l * DD * S3, bqkv + (size_t)l * S3,
            nullptr, nullptr, qkvb, S3, DD, 0);
        // V -> keys-contiguous layout for the P@V WMMA B-operand
        transpose_v<<<dim3((nvT + 255) / 256), 256, 0, stream>>>(qkvb, vT);
        // flash attention -> attb (bf16)
        attn_kernel<<<dim3(TT / 16, BB * HH), 32, 0, stream>>>(qkvb, vT, attb);
        // h = h + attb @ Wproj + bproj
        gemm_bf16_wmma<4><<<dim3(DD / 256, MM / 16), gemmBlk, 0, stream>>>(
            attb, WprojP + (size_t)l * DD * DD, bproj + (size_t)l * DD,
            h, h, nullptr, DD, DD, 0);
        // x = LN2(h) -> bf16
        ln_kernel<<<dim3(MM), 256, 0, stream>>>(h, ln2_g + l * DD, ln2_b + l * DD, xb, nullptr, DD);
        // ff = gelu(x @ W1 + b1) (bf16 out)
        gemm_bf16_wmma<4><<<dim3(FFN / 256, MM / 16), gemmBlk, 0, stream>>>(
            xb, W1P + (size_t)l * DD * FFN, b1 + (size_t)l * FFN,
            nullptr, nullptr, ffb, FFN, DD, 1);
        // h = h + ff @ W2 + b2
        gemm_bf16_wmma<4><<<dim3(DD / 256, MM / 16), gemmBlk, 0, stream>>>(
            ffb, W2P + (size_t)l * FFN * DD, b2 + (size_t)l * DD,
            h, h, nullptr, DD, FFN, 0);
    }

    // outputs: [next_event (M*IN) | action (B*ACT) | h (M*D)]
    float* out_ne  = (float*)d_out;
    float* out_act = out_ne + (size_t)MM * INN;
    float* out_h   = out_act + (size_t)BB * ACTN;

    // final LN -> f32 into d_out h-section + bf16 for head GEMM
    ln_kernel<<<dim3(MM), 256, 0, stream>>>(h, lnf_g, lnf_b, xb, out_h, DD);
    // next_event = hf @ W_ne + b_ne  (N=32 -> 2 tiles, single wave per block-row)
    gemm_bf16_wmma<2><<<dim3(1, MM / 16), dim3(32), 0, stream>>>(
        xb, WneP, b_ne, nullptr, out_ne, nullptr, INN, DD, 0);
    // action head
    action_kernel<<<dim3(1), 32, 0, stream>>>(out_h, W_act, b_act, out_act);
}
